// RPNLoss_32255204393371
// MI455X (gfx1250) — compile-verified, matrix operand order checked by reading, not run on hardware
//
#include <hip/hip_runtime.h>
#include <stdint.h>

typedef __attribute__((ext_vector_type(16))) _Float16 v16h;
typedef __attribute__((ext_vector_type(8)))  float    v8f;
typedef __attribute__((ext_vector_type(2)))  float    v2f;
typedef __attribute__((ext_vector_type(4)))  unsigned v4u;
typedef __attribute__((ext_vector_type(8)))  unsigned v8u;

typedef __attribute__((address_space(1))) int glob_int_t;
typedef __attribute__((address_space(3))) int lds_int_t;

#define A_TOT 36864
#define B_IMG 8
#define G_GT  128
#define HW_SZ 4096
#define NBLK3 9          // 9 blocks * 256 threads * 16 anchors = 36864

// ---- async global->LDS staging (ASYNCcnt path), guarded for portability ----
#if __has_builtin(__builtin_amdgcn_global_load_async_to_lds_b32) && \
    __has_builtin(__builtin_amdgcn_s_wait_asynccnt)
#define USE_ASYNC_LDS 1
__device__ __forceinline__ void async_ld_b32(const float* gsrc, float* ldst) {
  __builtin_amdgcn_global_load_async_to_lds_b32(
      (glob_int_t*)gsrc, (lds_int_t*)ldst, 0, 0);
}
#endif

// ---- Tensor Data Mover: DMA one contiguous row of `ndw` dwords into LDS ----
// D# per CDNA5 ISA ch.8: group0 = {count, lds_addr, global_addr, type=2},
// group1 = {data_size=4B, tensor/tile dim0 = ndw, dim1 = 1, stride = ndw}.
// 2-group form (VADDR2/3 = NULL) is valid for tensors up to 2D.
__device__ __forceinline__ void tdm_load_row(const float* gsrc, unsigned lds_off,
                                             unsigned ndw) {
  unsigned long long ga = (unsigned long long)(uintptr_t)gsrc;
  v4u g0;
  g0[0] = 1u;                                                // count=1 (valid user D#)
  g0[1] = lds_off;                                           // lds_addr (bytes)
  g0[2] = (unsigned)ga;                                      // global_addr[31:0]
  g0[3] = (unsigned)((ga >> 32) & 0x01FFFFFFull) | (2u << 30); // addr[56:32] | type=2
  v8u g1;
  g1[0] = 0x20000u;                 // workgroup_mask=0, data_size=2 (4 bytes)
  g1[1] = (ndw & 0xFFFFu) << 16;    // tensor_dim0[15:0]
  g1[2] = (ndw >> 16) | (1u << 16); // tensor_dim0[31:16] | tensor_dim1=1
  g1[3] = (ndw & 0xFFFFu) << 16;    // tile_dim0
  g1[4] = 1u;                       // tile_dim1=1, tile_dim2=0
  g1[5] = ndw;                      // tensor_dim0_stride[31:0]
  g1[6] = 0u;                       // stride hi, tensor_dim1_stride lo
  g1[7] = 0u;
  asm volatile("tensor_load_to_lds %0, %1" :: "s"(g0), "s"(g1) : "memory");
}

// ---------------- helpers ----------------

__device__ __forceinline__ unsigned hash32(unsigned x) {
  x ^= x >> 16; x *= 0x7feb352dU;
  x ^= x >> 15; x *= 0x846ca68bU;
  x ^= x >> 16;
  return x;
}

__device__ __forceinline__ float bperm_f(float v, int srcLane) {
  return __int_as_float(__builtin_amdgcn_ds_bpermute(srcLane << 2, __float_as_int(v)));
}

// WMMA f16 wave reduction: 16 f16 values/lane (512 total), exact for 0/1 flags
// (f32 accumulate). One xor-16 bpermute folds the two row-halves.
__device__ __forceinline__ float wave_wmma_sum_f16(v16h aVals) {
  v16h ones;
  #pragma unroll
  for (int i = 0; i < 16; ++i) ones[i] = (_Float16)1.0f;
  v8f c = {};
  v8f d = __builtin_amdgcn_wmma_f32_16x16x32_f16(
      false, aVals, false, ones, (short)0, c, false, false);
  float s = d[0] + d[1] + d[2] + d[3] + d[4] + d[5] + d[6] + d[7];
  int lane = threadIdx.x & 31;
  s += bperm_f(s, lane ^ 16);
  return s;
}

// WMMA f32 wave reduction via V_WMMA_F32_16X16X4_F32:
// A (16x4) = lane scalar in K-slot, B = ones -> D rows replicate row sums;
// summing the 8 D regs gives rows 0-7 (lanes 0-15) / rows 8-15 (lanes 16-31),
// one xor-16 bpermute yields the full-precision wave total (fixed order).
__device__ __forceinline__ float wave_wmma_sum_f32(float v) {
  v2f a; a[0] = v;    a[1] = 0.0f;
  v2f bm; bm[0] = 1.0f; bm[1] = 1.0f;
  v8f c = {};
  v8f d = __builtin_amdgcn_wmma_f32_16x16x4_f32(
      false, a, false, bm, (short)0, c, false, false);
  float s = d[0] + d[1] + d[2] + d[3] + d[4] + d[5] + d[6] + d[7];
  int lane = threadIdx.x & 31;
  s += bperm_f(s, lane ^ 16);
  return s;
}

__device__ __forceinline__ float iou_box(float ax0, float ay0, float ax1, float ay1,
                                         float gx0, float gy0, float gx1, float gy1) {
  float lx = fmaxf(ax0, gx0), ly = fmaxf(ay0, gy0);
  float rx = fminf(ax1, gx1), ry = fminf(ay1, gy1);
  float w = fmaxf(rx - lx, 0.0f), h = fmaxf(ry - ly, 0.0f);
  float inter = w * h;
  float aa = (ax1 - ax0) * (ay1 - ay0);
  float ag = (gx1 - gx0) * (gy1 - gy0);
  return inter / (aa + ag - inter + 1e-6f);
}

__device__ __forceinline__ float smooth_l1(float d) {
  float ad = fabsf(d);
  return ad < 1.0f ? 0.5f * d * d : ad - 0.5f;
}

// ---------------- kernels ----------------

// zero the scatter/accumulator regions each call (ws is poisoned by harness)
__global__ void k0_init(unsigned* __restrict__ forced, unsigned* __restrict__ counts) {
  int i = blockIdx.x * 256 + threadIdx.x;
  if (i < B_IMG * A_TOT) forced[i] = 0u;
  if (i < B_IMG * 4)     counts[i] = 0u;
}

// per-anchor: max IoU over 128 gts + argmax. GT tile staged via TDM
// (tensor_load_to_lds, TENSORcnt) issued once by wave 0.
__global__ void k1_anchor_best(const float* __restrict__ anchors,
                               const float* __restrict__ gt,
                               unsigned long long* __restrict__ packed) {
  __shared__ float sgt[G_GT * 4];
  int b = blockIdx.y;
  int t = threadIdx.x;
  if (t < 32) {
    tdm_load_row(gt + b * 512, (unsigned)(uintptr_t)(lds_int_t*)sgt, 512u);
    asm volatile("s_wait_tensorcnt 0x0" ::: "memory");
  }
  __syncthreads();

  int a = blockIdx.x * 256 + t;
  float4 ab = reinterpret_cast<const float4*>(anchors)[a];
  float best = -1.0f;
  int bg = 0;
  for (int g = 0; g < G_GT; ++g) {
    float gx0 = sgt[g * 4 + 0], gy0 = sgt[g * 4 + 1];
    float gx1 = sgt[g * 4 + 2], gy1 = sgt[g * 4 + 3];
    float iou = iou_box(ab.x, ab.y, ab.z, ab.w, gx0, gy0, gx1, gy1);
    if (iou > best) { best = iou; bg = g; }   // ties -> first (matches argmax)
  }
  packed[b * A_TOT + a] =
      ((unsigned long long)__float_as_uint(best) << 32) | (unsigned)bg;
}

// per-gt: argmax IoU over all anchors (transposed pass, anchors L2-resident),
// scatter "forced positive" flag. Tie-break toward smallest anchor via ~a key.
__global__ void k2_gt_best(const float* __restrict__ anchors,
                           const float* __restrict__ gt,
                           unsigned* __restrict__ forced) {
  __shared__ unsigned long long red[256];
  int b = blockIdx.y, g = blockIdx.x, t = threadIdx.x;
  float gx0 = gt[b * 512 + g * 4 + 0];
  float gy0 = gt[b * 512 + g * 4 + 1];
  float gx1 = gt[b * 512 + g * 4 + 2];
  float gy1 = gt[b * 512 + g * 4 + 3];

  unsigned long long best = 0ull;
  for (int a = t; a < A_TOT; a += 256) {
    if (a + 256 < A_TOT)
      __builtin_prefetch(anchors + 4 * (a + 256), 0, 1);   // global_prefetch
    float4 ab = reinterpret_cast<const float4*>(anchors)[a];
    float iou = iou_box(ab.x, ab.y, ab.z, ab.w, gx0, gy0, gx1, gy1);
    unsigned long long pk =
        ((unsigned long long)__float_as_uint(iou) << 32) | (unsigned)(~(unsigned)a);
    if (pk > best) best = pk;
  }
  red[t] = best;
  __syncthreads();
  for (int s = 128; s > 0; s >>= 1) {
    if (t < s) { if (red[t + s] > red[t]) red[t] = red[t + s]; }
    __syncthreads();
  }
  if (t == 0) {
    unsigned a = ~(unsigned)(red[0] & 0xffffffffu);
    forced[b * A_TOT + a] = 1u;   // benign multi-gt overlap: all write 1
  }
}

// count pos0/neg0 per image; 16 anchors/thread, flags reduced via f16 WMMA,
// integer atomics (deterministic) into per-image counters.
__global__ void k3_counts(const unsigned long long* __restrict__ packed,
                          const unsigned* __restrict__ forced,
                          unsigned* __restrict__ counts) {
  int b = blockIdx.y, t = threadIdx.x;
  int base = blockIdx.x * 4096;
  v16h fpos = {}, fneg = {};
  #pragma unroll
  for (int i = 0; i < 16; ++i) {
    int a = base + i * 256 + t;
    unsigned long long pk = packed[b * A_TOT + a];
    float iou = __uint_as_float((unsigned)(pk >> 32));
    bool f = forced[b * A_TOT + a] != 0u;
    bool pos0 = (iou >= 0.7f) || f;
    bool neg0 = (iou < 0.3f) && !f;
    fpos[i] = pos0 ? (_Float16)1.0f : (_Float16)0.0f;
    fneg[i] = neg0 ? (_Float16)1.0f : (_Float16)0.0f;
  }
  float np = wave_wmma_sum_f16(fpos);   // exact: 0/1 in f16, f32 accumulate
  float nn = wave_wmma_sum_f16(fneg);
  if ((t & 31) == 0) {
    atomicAdd(&counts[b * 4 + 0], (unsigned)(np + 0.5f));
    atomicAdd(&counts[b * 4 + 1], (unsigned)(nn + 0.5f));
  }
}

// derive sampling targets per image
__global__ void k4_targets(unsigned* __restrict__ counts) {
  int b = threadIdx.x;
  if (b < B_IMG) {
    unsigned np = counts[b * 4 + 0];
    unsigned nn = counts[b * 4 + 1];
    unsigned tp = np < 128u ? np : 128u;
    unsigned rem = 256u - tp;
    unsigned tn = nn < rem ? nn : rem;
    counts[b * 4 + 2] = tp;
    counts[b * 4 + 3] = tn;
  }
}

// fused selection + BCE + SmoothL1; all four sums reduced through the f32
// matrix pipe (V_WMMA_F32_16X16X4_F32); block partials written in fixed order.
// GT tile staged via per-lane async copies (ASYNCcnt path).
__global__ void k5_loss(const float* __restrict__ cls,
                        const float* __restrict__ boxp,
                        const float* __restrict__ anchors,
                        const float* __restrict__ gt,
                        const unsigned long long* __restrict__ packed,
                        const unsigned* __restrict__ forced,
                        const unsigned* __restrict__ counts,
                        float* __restrict__ partials) {
  __shared__ float sgt[G_GT * 4];
  __shared__ float wsum[8][4];
  int b = blockIdx.y, t = threadIdx.x;
#ifdef USE_ASYNC_LDS
  async_ld_b32(gt + b * 512 + t,       &sgt[t]);
  async_ld_b32(gt + b * 512 + t + 256, &sgt[t + 256]);
  __builtin_amdgcn_s_wait_asynccnt(0);
#else
  sgt[t]       = gt[b * 512 + t];
  sgt[t + 256] = gt[b * 512 + t + 256];
#endif
  __syncthreads();

  unsigned np = counts[b * 4 + 0], nn = counts[b * 4 + 1];
  unsigned long long thp = (unsigned long long)counts[b * 4 + 2] << 32;
  unsigned long long thn = (unsigned long long)counts[b * 4 + 3] << 32;

  int base = blockIdx.x * 4096;
  float sb = 0.0f, sl = 0.0f, cv = 0.0f, cp = 0.0f;
  #pragma unroll
  for (int i = 0; i < 16; ++i) {
    int a = base + i * 256 + t;
    unsigned long long pk = packed[b * A_TOT + a];
    float iou = __uint_as_float((unsigned)(pk >> 32));
    int g = (int)(pk & 0xffffffffu);
    bool f = forced[b * A_TOT + a] != 0u;
    bool pos0 = (iou >= 0.7f) || f;
    bool neg0 = (iou < 0.3f) && !f;

    unsigned key = (unsigned)(b * A_TOT + a);
    unsigned hp = hash32(key * 2654435761u + 0x1234567u);
    unsigned hn = hash32(key * 2246822519u + 0x89abcdeu);
    // keep iff h/2^32 < target/num  (calibrated Bernoulli; exact when target==num)
    bool keep_pos = pos0 && ((unsigned long long)hp * (unsigned long long)np < thp);
    bool keep_neg = neg0 && ((unsigned long long)hn * (unsigned long long)nn < thn);
    bool valid = keep_pos || keep_neg;

    int hw = a / 9, k = a - hw * 9;          // a = (h*W+w)*9 + k
    if (valid) {
      float x = cls[(b * 9 + k) * HW_SZ + hw];
      float z = keep_pos ? 1.0f : 0.0f;
      sb += fmaxf(x, 0.0f) - x * z + log1pf(expf(-fabsf(x)));   // stable BCE
    }
    if (keep_pos) {
      float4 ab = reinterpret_cast<const float4*>(anchors)[a];
      float gx0 = sgt[g * 4 + 0], gy0 = sgt[g * 4 + 1];
      float gx1 = sgt[g * 4 + 2], gy1 = sgt[g * 4 + 3];
      float aw = ab.z - ab.x, ah = ab.w - ab.y;
      float acx = 0.5f * (ab.x + ab.z), acy = 0.5f * (ab.y + ab.w);
      float gw = gx1 - gx0, gh = gy1 - gy0;
      float gcx = 0.5f * (gx0 + gx1), gcy = 0.5f * (gy0 + gy1);
      float t0 = (gcx - acx) / aw, t1 = (gcy - acy) / ah;
      float t2 = logf(gw / aw),    t3 = logf(gh / ah);
      int boff = (b * 36 + k * 4) * HW_SZ + hw;
      float p0 = boxp[boff];
      float p1 = boxp[boff + HW_SZ];
      float p2 = boxp[boff + 2 * HW_SZ];
      float p3 = boxp[boff + 3 * HW_SZ];
      sl += smooth_l1(p0 - t0) + smooth_l1(p1 - t1) +
            smooth_l1(p2 - t2) + smooth_l1(p3 - t3);
    }
    cv += valid    ? 1.0f : 0.0f;
    cp += keep_pos ? 1.0f : 0.0f;
  }

  float sbw = wave_wmma_sum_f32(sb);
  float slw = wave_wmma_sum_f32(sl);
  float nvw = wave_wmma_sum_f32(cv);
  float npw = wave_wmma_sum_f32(cp);

  int wave = t >> 5;
  if ((t & 31) == 0) {
    wsum[wave][0] = sbw; wsum[wave][1] = nvw;
    wsum[wave][2] = slw; wsum[wave][3] = npw;
  }
  __syncthreads();
  if (t == 0) {
    float a0 = 0, a1 = 0, a2 = 0, a3 = 0;
    for (int w = 0; w < 8; ++w) {          // fixed order -> deterministic
      a0 += wsum[w][0]; a1 += wsum[w][1];
      a2 += wsum[w][2]; a3 += wsum[w][3];
    }
    float* p = &partials[(b * NBLK3 + blockIdx.x) * 4];
    p[0] = a0; p[1] = a1; p[2] = a2; p[3] = a3;
  }
}

// final fixed-order combine -> 3 outputs
__global__ void k6_final(const float* __restrict__ partials, float* __restrict__ out) {
  if (threadIdx.x == 0 && blockIdx.x == 0) {
    float clsm = 0.0f, boxm = 0.0f;
    for (int b = 0; b < B_IMG; ++b) {
      float sb = 0, sv = 0, sl = 0, sp = 0;
      for (int blk = 0; blk < NBLK3; ++blk) {
        const float* p = &partials[(b * NBLK3 + blk) * 4];
        sb += p[0]; sv += p[1]; sl += p[2]; sp += p[3];
      }
      clsm += sb / fmaxf(sv, 1.0f);
      boxm += sl / fmaxf(4.0f * sp, 1.0f);
    }
    clsm *= (1.0f / 8.0f);
    boxm *= (1.0f / 8.0f);
    out[0] = clsm; out[1] = boxm; out[2] = clsm + boxm;
  }
}

// ---------------- launcher ----------------

extern "C" void kernel_launch(void* const* d_in, const int* in_sizes, int n_in,
                              void* d_out, int out_size, void* d_ws, size_t ws_size,
                              hipStream_t stream) {
  const float* cls     = (const float*)d_in[0];   // [8,9,64,64]
  const float* boxp    = (const float*)d_in[1];   // [8,36,64,64]
  const float* anchors = (const float*)d_in[2];   // [36864,4]
  const float* gt      = (const float*)d_in[3];   // [8,128,4]
  float* out = (float*)d_out;
  (void)in_sizes; (void)n_in; (void)out_size; (void)ws_size;

  char* ws = (char*)d_ws;
  unsigned long long* packed = (unsigned long long*)ws;              // 2,359,296 B
  unsigned* forced   = (unsigned*)(ws + 2359296);                    // 1,179,648 B
  unsigned* counts   = (unsigned*)(ws + 2359296 + 1179648);          //       128 B
  float*    partials = (float*)   (ws + 2359296 + 1179648 + 256);    //     1,152 B

  k0_init      <<<dim3((B_IMG * A_TOT + 255) / 256), 256, 0, stream>>>(forced, counts);
  k1_anchor_best<<<dim3(A_TOT / 256, B_IMG), 256, 0, stream>>>(anchors, gt, packed);
  k2_gt_best   <<<dim3(G_GT, B_IMG),        256, 0, stream>>>(anchors, gt, forced);
  k3_counts    <<<dim3(NBLK3, B_IMG),       256, 0, stream>>>(packed, forced, counts);
  k4_targets   <<<dim3(1),                   32, 0, stream>>>(counts);
  k5_loss      <<<dim3(NBLK3, B_IMG),       256, 0, stream>>>(cls, boxp, anchors, gt,
                                                              packed, forced, counts, partials);
  k6_final     <<<dim3(1),                   32, 0, stream>>>(partials, out);
}
